// EmbeddingModule_84550726189764
// MI455X (gfx1250) — matrix operand order repaired and checked by reading, compiled, not run on hardware
//
#include <hip/hip_runtime.h>
#include <hip/hip_bf16.h>

// ---------------------------------------------------------------------------
// Problem constants (from reference): B=1024, D=128, N=100000, HIST=50, k=100
// k' = 150. N = 25 chunks x 4000 items; 4000 = 250 tiles of 16.
// ---------------------------------------------------------------------------
#define BQ      1024
#define DIM     128
#define NITEMS  100000
#define HIST    50
#define TOPK    100
#define KPRIME  150
#define CHUNK   4000
#define NCHUNK  25
#define CAND    160            // per-chunk per-row candidates (>= KPRIME)
#define NTILES  (CHUNK / 16)   // 250
#define WAVES   16             // 512-thread block: 4 waves per SIMD32
#define NEG_INF (-3.402823466e38f)

typedef __attribute__((ext_vector_type(16))) __bf16 v16bf;
typedef __attribute__((ext_vector_type(8)))  __bf16 v8bf;
typedef __attribute__((ext_vector_type(8)))  float  v8f;

// ---------------------------------------------------------------------------
// Kernel 0a: convert Q (f32, row-major [B][D]) -> bf16 [B][D]
// ---------------------------------------------------------------------------
__global__ void convert_q_kernel(const float* __restrict__ q,
                                 __bf16* __restrict__ qbf, int n) {
  int i = blockIdx.x * blockDim.x + threadIdx.x;
  if (i < n) qbf[i] = (__bf16)q[i];
}

// ---------------------------------------------------------------------------
// Kernel 0b: transpose+convert E_t (f32 [D][N]) -> Ebf (bf16 [N][D]).
// 32x32 LDS tile transpose; coalesced both sides. Done once; E then lives in
// the 192MB L2 for the whole pipeline.
// ---------------------------------------------------------------------------
__global__ void transpose_e_kernel(const float* __restrict__ et,
                                   __bf16* __restrict__ ebf) {
  __shared__ float tile[32][33];
  const int nt = blockIdx.x * 32;   // item tile (3125 blocks, exact)
  const int dt = blockIdx.y * 32;   // dim tile  (4 blocks)
  const int tx = threadIdx.x;       // 0..31
  const int ty = threadIdx.y;       // 0..7
#pragma unroll
  for (int j = 0; j < 4; ++j) {
    int d = dt + ty + j * 8;
    int n = nt + tx;
    tile[ty + j * 8][tx] = et[(size_t)d * NITEMS + n];
  }
  __syncthreads();
#pragma unroll
  for (int j = 0; j < 4; ++j) {
    int n = nt + ty + j * 8;
    ebf[(size_t)n * DIM + dt + tx] = (__bf16)tile[tx][ty + j * 8];
  }
}

// ---------------------------------------------------------------------------
// Kernel 1: fused WMMA GEMM (16 rows x 4000-item chunk) + exact per-row
// top-160 selection in LDS. Grid: (B/16, NCHUNK). Block: 512 (16 waves).
// Dynamic LDS: 16*4000*4 = 256000 bytes (CDNA5's 320KB WGP LDS).
// Ping-pong double-buffered B-fragment loads (2x unrolled, no register
// rotation) hide L2 latency under the WMMAs.
// ---------------------------------------------------------------------------
__global__ void score_select_kernel(const __bf16* __restrict__ qbf,
                                    const __bf16* __restrict__ ebf,
                                    float* __restrict__ cand_score,
                                    int*   __restrict__ cand_id) {
  extern __shared__ float smem[];
  float (*s_sc)[CHUNK] = (float(*)[CHUNK])smem;   // [16][4000]

  const int lane  = threadIdx.x & 31;
  const int wave  = threadIdx.x >> 5;             // 0..15
  const int half  = lane >> 4;                    // 0 or 1
  const int l16   = lane & 15;
  const int rowBase   = blockIdx.x * 16;
  const int chunk     = blockIdx.y;
  const int chunkBase = chunk * CHUNK;

  // --- A fragments: 16x32 bf16 per K-step, row m = l16, in this code's k order:
  // elements 0..15 -> k = 16*half + 0..15 (contiguous, matches B)
  v16bf afrag[4];
  {
    const __bf16* qp = qbf + (size_t)(rowBase + l16) * DIM;
#pragma unroll
    for (int kk = 0; kk < 4; ++kk) {
      const v8bf a0 = *(const v8bf*)(qp + kk * 32 + 16 * half);
      const v8bf a1 = *(const v8bf*)(qp + kk * 32 + 16 * half + 8);
#pragma unroll
      for (int i = 0; i < 8; ++i) {
        afrag[kk][i]     = a0[i];
        afrag[kk][i + 8] = a1[i];
      }
    }
  }

  // B fragment pointer for tile t: column n = l16, k = kk*32 + 16*half + 0..15
  // -> one aligned 32-byte contiguous run per kk in Ebf[n][.]
  auto bptr = [&](int t) {
    return ebf + (size_t)(chunkBase + t * 16 + l16) * DIM + 16 * half;
  };
  auto loadB = [&](v16bf (&buf)[4], int t) {
#pragma unroll
    for (int kk = 0; kk < 4; ++kk)
      buf[kk] = *(const v16bf*)(bptr(t) + kk * 32);
  };
  auto computeTile = [&](const v16bf (&buf)[4], int t) {
    v8f c = {};
#pragma unroll
    for (int kk = 0; kk < 4; ++kk) {
      c = __builtin_amdgcn_wmma_f32_16x16x32_bf16(
          /*neg_a=*/false, afrag[kk], /*neg_b=*/false, buf[kk],
          /*c_mod=*/(short)0, c, /*reuse_a=*/false, /*reuse_b=*/false);
    }
    // C layout: VGPR r -> m = r + 8*half, n = l16
#pragma unroll
    for (int r = 0; r < 8; ++r)
      s_sc[r + 8 * half][t * 16 + l16] = c[r];
  };

  // --- GEMM over 250 tiles, 16 waves round-robin, ping-pong B buffers ---
  v16bf buf0[4], buf1[4];
  if (wave < NTILES) loadB(buf0, wave);
  for (int t = wave; t < NTILES; t += 2 * WAVES) {
    const int tA = t + WAVES;
    if (tA < NTILES) {
      loadB(buf1, tA);
      __builtin_prefetch(bptr(tA) + (size_t)(16 * WAVES) * DIM, 0, 0);
    }
    computeTile(buf0, t);
    if (tA < NTILES) {
      const int tB = tA + WAVES;
      if (tB < NTILES) loadB(buf0, tB);
      computeTile(buf1, tA);
    }
  }
  __syncthreads();

  // --- Exact per-row top-CAND: 1 row per wave, butterfly argmax.
  // All lanes converge on the same (score,index) winner (index tie-break),
  // so no block barrier is needed: same-wave DS ops retire in order.
  {
    const int m = wave;   // row within tile, 16 waves <-> 16 rows
    for (int it = 0; it < CAND; ++it) {
      float best = NEG_INF;
      int   bi   = 0;
      for (int i = lane; i < CHUNK; i += 32) {
        const float s = s_sc[m][i];
        if (s > best) { best = s; bi = i; }
      }
#pragma unroll
      for (int off = 16; off > 0; off >>= 1) {
        const float ov = __shfl_xor(best, off, 32);
        const int   oi = __shfl_xor(bi,   off, 32);
        if (ov > best || (ov == best && oi < bi)) { best = ov; bi = oi; }
      }
      if (lane == 0) {
        const size_t o = ((size_t)(rowBase + m) * NCHUNK + chunk) * CAND + it;
        cand_score[o] = best;
        cand_id[o]    = chunkBase + bi;
        s_sc[m][bi]   = NEG_INF;   // in-order vs this wave's next DS loads
      }
    }
  }
}

// ---------------------------------------------------------------------------
// Kernel 2: per-row merge of 25*160 = 4000 candidates -> sorted top-150,
// invalid-id filter (stable, valid-first), emit top-100 ids (as float) and
// scores, zero-padded. One block (256 threads) per row.
// ---------------------------------------------------------------------------
__global__ void merge_filter_kernel(const float* __restrict__ cand_score,
                                    const int*   __restrict__ cand_id,
                                    const int*   __restrict__ item_ids,
                                    const int*   __restrict__ invalid_ids,
                                    float* __restrict__ out) {
  __shared__ float cs[NCHUNK * CAND];
  __shared__ int   ci[NCHUNK * CAND];
  __shared__ float rv[256];
  __shared__ int   ri[256];
  __shared__ float sel_s[KPRIME];
  __shared__ int   sel_i[KPRIME];
  __shared__ int   inv[HIST];

  const int row = blockIdx.x;
  const int tid = threadIdx.x;
  const int NC  = NCHUNK * CAND;

  for (int i = tid; i < NC; i += 256) {
    cs[i] = cand_score[(size_t)row * NC + i];
    ci[i] = cand_id[(size_t)row * NC + i];
  }
  if (tid < HIST) inv[tid] = invalid_ids[row * HIST + tid];
  __syncthreads();

  // KPRIME iterations of block-wide argmax -> descending-sorted top-150
  for (int it = 0; it < KPRIME; ++it) {
    float best = NEG_INF;
    int   bi   = 0;
    for (int i = tid; i < NC; i += 256) {
      const float s = cs[i];
      if (s > best) { best = s; bi = i; }
    }
    rv[tid] = best; ri[tid] = bi;
    __syncthreads();
#pragma unroll
    for (int s = 128; s > 0; s >>= 1) {
      if (tid < s && rv[tid + s] > rv[tid]) {
        rv[tid] = rv[tid + s]; ri[tid] = ri[tid + s];
      }
      __syncthreads();
    }
    if (tid == 0) {
      sel_s[it]  = rv[0];
      sel_i[it]  = ci[ri[0]];
      cs[ri[0]]  = NEG_INF;
    }
    __syncthreads();
  }

  // Stable valid-first filter + zero pad (serial, 150x50 compares)
  if (tid == 0) {
    float* out_ids    = out;
    float* out_scores = out + (size_t)BQ * TOPK;
    int cnt = 0;
    for (int j = 0; j < KPRIME && cnt < TOPK; ++j) {
      const int id = item_ids[sel_i[j]];
      bool bad = false;
      for (int h = 0; h < HIST; ++h) bad = bad || (inv[h] == id);
      if (!bad) {
        out_ids[(size_t)row * TOPK + cnt]    = (float)id;
        out_scores[(size_t)row * TOPK + cnt] = sel_s[j];
        ++cnt;
      }
    }
    for (; cnt < TOPK; ++cnt) {
      out_ids[(size_t)row * TOPK + cnt]    = 0.0f;
      out_scores[(size_t)row * TOPK + cnt] = 0.0f;
    }
  }
}

// ---------------------------------------------------------------------------
// Launch: ws layout = [Qbf | Ebf | cand_score | cand_id]  (~58.6 MB)
// ---------------------------------------------------------------------------
extern "C" void kernel_launch(void* const* d_in, const int* in_sizes, int n_in,
                              void* d_out, int out_size, void* d_ws, size_t ws_size,
                              hipStream_t stream) {
  const float* q   = (const float*)d_in[0];   // [1024,128] f32
  const float* et  = (const float*)d_in[1];   // [128,100000] f32
  const int*   ids = (const int*)d_in[2];     // [100000]
  const int*   inv = (const int*)d_in[3];     // [1024,50]
  float* out = (float*)d_out;                 // ids then scores, flat

  char* ws = (char*)d_ws;
  const size_t qbf_bytes = (size_t)BQ * DIM * 2;           // 256 KB
  const size_t ebf_bytes = (size_t)NITEMS * DIM * 2;       // 25.6 MB
  const size_t cs_bytes  = (size_t)BQ * NCHUNK * CAND * 4; // 16.4 MB
  __bf16* qbf = (__bf16*)ws;
  __bf16* ebf = (__bf16*)(ws + qbf_bytes);
  float*  csc = (float*)(ws + qbf_bytes + ebf_bytes);
  int*    cid = (int*)(ws + qbf_bytes + ebf_bytes + cs_bytes);

  convert_q_kernel<<<(BQ * DIM + 255) / 256, 256, 0, stream>>>(q, qbf, BQ * DIM);
  transpose_e_kernel<<<dim3(NITEMS / 32, DIM / 32), dim3(32, 8), 0, stream>>>(et, ebf);
  score_select_kernel<<<dim3(BQ / 16, NCHUNK), 32 * WAVES,
                        16 * CHUNK * sizeof(float), stream>>>(qbf, ebf, csc, cid);
  merge_filter_kernel<<<BQ, 256, 0, stream>>>(csc, cid, ids, inv, out);
}
